// WindowAttention_58136677318672
// MI455X (gfx1250) — matrix operand order repaired and checked by reading, not compiled, not run on hardware
//
#include <hip/hip_runtime.h>
#include <hip/hip_bf16.h>
#include <math.h>

// ---------------------------------------------------------------------------
// Types for CDNA5 WMMA (wave32, v_wmma_f32_16x16x32_bf16)
// ---------------------------------------------------------------------------
typedef __bf16 bf16_t;
typedef __attribute__((ext_vector_type(16))) __bf16 v16bf;
typedef __attribute__((ext_vector_type(8)))  __bf16 v8bf;
typedef __attribute__((ext_vector_type(4)))  __bf16 v4bf;
typedef __attribute__((ext_vector_type(8)))  float  v8f;
typedef __attribute__((ext_vector_type(4)))  float  v4f;

__device__ __forceinline__ v8f zero_v8f() {
    v8f z;
#pragma unroll
    for (int j = 0; j < 8; ++j) z[j] = 0.0f;
    return z;
}

__device__ __forceinline__ v8f wmma_bf(v16bf a, v16bf b, v8f c) {
    // D = A(16x32 bf16) * B(32x16 bf16) + C(16x16 f32)
    return __builtin_amdgcn_wmma_f32_16x16x32_bf16(false, a, false, b,
                                                   (short)0, c, false, false);
}

// A-fragment 16x32 bf16.  rowptr -> element [row][k=0] of a 32-wide K slice
// (16B aligned).  Lane L<16: row=L, K 0..7 & 16..23 ; lane L>=16: K 8..15 & 24..31.
__device__ __forceinline__ v16bf frag_a(const bf16_t* rowptr, int lane) {
    int koff = (lane & 16) ? 8 : 0;
    v8bf lo = *(const v8bf*)(rowptr + koff);
    v8bf hi = *(const v8bf*)(rowptr + 16 + koff);
    return __builtin_shufflevector(lo, hi, 0,1,2,3,4,5,6,7,8,9,10,11,12,13,14,15);
}

// B-fragment 32x16 bf16 from K-major (transposed) storage: colptr -> [n][k=0]
// (16B aligned).  Lane L<16: col=L, K 0..7 & 16..23 ; lane L>=16: K 8..15 & 24..31.
__device__ __forceinline__ v16bf frag_b(const bf16_t* colptr, int lane) {
    int kb = (lane & 16) ? 8 : 0;      // 0 or 8
    v8bf lo = *(const v8bf*)(colptr + kb);
    v8bf hi = *(const v8bf*)(colptr + 16 + kb);
    return __builtin_shufflevector(lo, hi, 0,1,2,3,4,5,6,7,8,9,10,11,12,13,14,15);
}

__device__ __forceinline__ float gelu_f(float x) {
    return 0.5f * x * (1.0f + erff(x * 0.70710678118654752f));
}

// ---------------------------------------------------------------------------
// Kernel 1: fp32 -> bf16 weight conversion + transpose to N-major (N x K),
// so GEMM B-tiles stage as pure contiguous b128 copies (no LDS scatter).
// ---------------------------------------------------------------------------
__global__ void convert_weights(const float* __restrict__ wq, const float* __restrict__ wkv,
                                const float* __restrict__ pw, const float* __restrict__ hw,
                                bf16_t* __restrict__ owq, bf16_t* __restrict__ owkv,
                                bf16_t* __restrict__ opw, bf16_t* __restrict__ ohw) {
    int i = blockIdx.x * 256 + threadIdx.x;
    if (i < 65536)  { int k = i >> 8, n = i & 255; owq[n * 256 + k]  = (bf16_t)wq[i];  }  // 256x256
    if (i < 131072) { int k = i >> 9, n = i & 511; owkv[n * 256 + k] = (bf16_t)wkv[i]; }  // 256x512
    if (i < 65536)  { int k = i >> 8, n = i & 255; opw[n * 256 + k]  = (bf16_t)pw[i];  }  // 256x256
    if (i < 32768)  { int k = i >> 8, n = i & 255; ohw[n * 128 + k]  = (bf16_t)hw[i];  }  // 128x256
}

// ---------------------------------------------------------------------------
// Kernel 2: GEMM  out(bf16, MxN) = A(fp32, MxK) * Wt(bf16, NxK)^T + bias
// tile 128x128, 8 waves, double-buffered LDS software pipeline.
// ---------------------------------------------------------------------------
__global__ __launch_bounds__(256) void gemm_xw_f32a(
        const float* __restrict__ A, const bf16_t* __restrict__ Wt,
        const float* __restrict__ bias, bf16_t* __restrict__ out,
        int K, int N) {
    __shared__ bf16_t sA[2][128][40];   // 128 rows x 32 K
    __shared__ bf16_t sB[2][128][40];   // 128 cols x 32 K (K-major)
    int tid = threadIdx.x, lane = tid & 31, wv = tid >> 5;
    int laneM = lane & 15;
    int rowBase = blockIdx.x * 128;
    int c0 = blockIdx.y * 128;
    const int NK = K >> 5;

    auto loadA = [&](int k0, v4f aR[4]) {
#pragma unroll
        for (int it = 0; it < 4; ++it) {
            int g = tid + it * 256, r = g >> 3, f4 = g & 7;
            aR[it] = *(const v4f*)(A + (size_t)(rowBase + r) * K + k0 + f4 * 4);
        }
    };
    auto storeA = [&](int buf, const v4f aR[4]) {
#pragma unroll
        for (int it = 0; it < 4; ++it) {
            int g = tid + it * 256, r = g >> 3, f4 = g & 7;
            v4bf b4;
            b4[0] = (bf16_t)aR[it][0]; b4[1] = (bf16_t)aR[it][1];
            b4[2] = (bf16_t)aR[it][2]; b4[3] = (bf16_t)aR[it][3];
            *(v4bf*)&sA[buf][r][f4 * 4] = b4;
        }
    };
    auto loadB = [&](int k0, v8bf bR[2]) {
#pragma unroll
        for (int it = 0; it < 2; ++it) {
            int g = tid + it * 256, c = g >> 2, kg = g & 3;
            bR[it] = *(const v8bf*)(Wt + (size_t)(c0 + c) * K + k0 + kg * 8);
        }
    };
    auto storeB = [&](int buf, const v8bf bR[2]) {
#pragma unroll
        for (int it = 0; it < 2; ++it) {
            int g = tid + it * 256, c = g >> 2, kg = g & 3;
            *(v8bf*)&sB[buf][c][kg * 8] = bR[it];
        }
    };

    v8f acc[8];
#pragma unroll
    for (int nt = 0; nt < 8; ++nt) acc[nt] = zero_v8f();

    v4f aR[4]; v8bf bR[2];
    loadA(0, aR); loadB(0, bR);
    storeA(0, aR); storeB(0, bR);
    __syncthreads();
    for (int kk = 0; kk < NK; ++kk) {
        int cur = kk & 1;
        bool pre = (kk + 1 < NK);
        if (pre) { loadA((kk + 1) * 32, aR); loadB((kk + 1) * 32, bR); }
        v16bf af = frag_a(&sA[cur][wv * 16 + laneM][0], lane);
#pragma unroll
        for (int nt = 0; nt < 8; ++nt) {
            v16bf bfr = frag_b(&sB[cur][nt * 16 + laneM][0], lane);
            acc[nt] = wmma_bf(af, bfr, acc[nt]);
        }
        if (pre) { storeA(1 - cur, aR); storeB(1 - cur, bR); }
        __syncthreads();
    }

    int rsel = (lane & 16) ? 8 : 0;
#pragma unroll
    for (int nt = 0; nt < 8; ++nt) {
#pragma unroll
        for (int i = 0; i < 8; ++i) {
            int row = rowBase + wv * 16 + i + rsel;
            int col = c0 + nt * 16 + laneM;
            out[(size_t)row * N + col] = (bf16_t)(acc[nt][i] + bias[col]);
        }
    }
}

// ---------------------------------------------------------------------------
// Kernel 3: windowed flash attention, 1 wave = 1 head, 64-query row block,
// streaming 64-key blocks, online softmax, all matmuls on v_wmma bf16.
// ---------------------------------------------------------------------------
template <int WS, int SHIFT>
__global__ __launch_bounds__(128) void win_attn(
        const bf16_t* __restrict__ qb,   // 65536 x 256
        const bf16_t* __restrict__ kvb,  // 65536 x 512  (k | v)
        const float*  __restrict__ rpb,  // (2WS-1)^2 x 4
        bf16_t* __restrict__ xattn) {    // 65536 x 256
    constexpr int N   = WS * WS;
    constexpr int NRB = N / 64;          // row blocks per window
    constexpr int NWX = 64 / WS;
    constexpr int WPB = NWX * NWX;       // windows per batch image
    constexpr int G   = (WS == 8) ? 0 : 1;
    constexpr int NB  = 2 * WS - 1;

    __shared__ int           tokRow[N];
    __shared__ unsigned char tokReg[N];
    __shared__ float         sBias[NB * NB * 4];
    __shared__ bf16_t        pLds[4][64][72];
    __shared__ bf16_t        vtLds[4][32][72];

    int tid = threadIdx.x, lane = tid & 31, h = tid >> 5;
    int laneM = lane & 15;
    int rsel = (lane & 16) ? 8 : 0;
    int wg = blockIdx.x;
    int rb = wg % NRB;
    int win = wg / NRB;
    int b = win / WPB, w = win % WPB;
    int wh = w / NWX, ww = w % NWX;

    for (int t = tid; t < N; t += 128) {
        int ly = t / WS, lx = t - ly * WS;
        int gy = wh * WS + ly, gx = ww * WS + lx;        // unshifted coords (mask)
        int hy = (gy + SHIFT) & 63, hx = (gx + SHIFT) & 63;
        tokRow[t] = (b << 12) | (hy << 6) | hx;
        int ry = (gy < 64 - WS) ? 0 : ((gy < 64 - SHIFT) ? 1 : 2);
        int rx = (gx < 64 - WS) ? 0 : ((gx < 64 - SHIFT) ? 1 : 2);
        tokReg[t] = (unsigned char)(ry * 3 + rx);
    }
    for (int i = tid; i < NB * NB * 4; i += 128) sBias[i] = rpb[i];
    __syncthreads();

    const int q0 = rb * 64;
    const int cq = G * 128 + h * 32;     // q/k channel base
    const int cv = 256 + cq;             // v channel base inside kv
    const float scale = 0.17677669529663687f;   // 32^-0.5

    // Q A-fragments, loaded once (direct global, per-lane contiguous)
    v16bf aQ[4];
#pragma unroll
    for (int ti = 0; ti < 4; ++ti) {
        int tok = q0 + ti * 16 + laneM;
        aQ[ti] = frag_a(qb + (size_t)tokRow[tok] * 256 + cq, lane);
    }

    float mrun[4][8], lrun[4][8];
    v8f o[4][2];
#pragma unroll
    for (int ti = 0; ti < 4; ++ti) {
#pragma unroll
        for (int i = 0; i < 8; ++i) { mrun[ti][i] = -1e30f; lrun[ti][i] = 0.0f; }
        o[ti][0] = zero_v8f(); o[ti][1] = zero_v8f();
    }

    for (int kb = 0; kb < NRB; ++kb) {
        // stage V^T (32 ch x 64 tokens) for this wave's head
        for (int t = lane; t < 64; t += 32) {
            const bf16_t* vp = kvb + (size_t)tokRow[kb * 64 + t] * 512 + cv;
#pragma unroll
            for (int cc = 0; cc < 32; cc += 8) {
                v8bf vv = *(const v8bf*)(vp + cc);
#pragma unroll
                for (int j = 0; j < 8; ++j) vtLds[h][cc + j][t] = vv[j];
            }
        }
        // S = Q K^T  (16 wmma, K=32 == head dim -> single exact K-step)
        v8f s[4][4];
#pragma unroll
        for (int tj = 0; tj < 4; ++tj) {
            int tok = kb * 64 + tj * 16 + laneM;
            v16bf bK = frag_b(kvb + (size_t)tokRow[tok] * 512 + cq, lane);
#pragma unroll
            for (int ti = 0; ti < 4; ++ti) s[ti][tj] = wmma_bf(aQ[ti], bK, zero_v8f());
        }
        // scale + relative-position bias + shift mask
#pragma unroll
        for (int ti = 0; ti < 4; ++ti) {
#pragma unroll
            for (int i = 0; i < 8; ++i) {
                int r = q0 + ti * 16 + i + rsel;
                int rly = r / WS, rlx = r - rly * WS;
                int rreg = tokReg[r];
#pragma unroll
                for (int tj = 0; tj < 4; ++tj) {
                    int c = kb * 64 + tj * 16 + laneM;
                    int cly = c / WS, clx = c - cly * WS;
                    int bidx = ((rly - cly + WS - 1) * NB + (rlx - clx + WS - 1)) * 4 + h;
                    float v = s[ti][tj][i] * scale + sBias[bidx];
                    if (tokReg[c] != rreg) v -= 100.0f;
                    s[ti][tj][i] = v;
                }
            }
        }
        // online softmax; row stats uniform across 16-lane halves (wave32)
#pragma unroll
        for (int ti = 0; ti < 4; ++ti) {
            float corr[8];
#pragma unroll
            for (int i = 0; i < 8; ++i) {
                float m = s[ti][0][i];
                m = fmaxf(m, s[ti][1][i]); m = fmaxf(m, s[ti][2][i]); m = fmaxf(m, s[ti][3][i]);
                m = fmaxf(m, __shfl_xor(m, 1, 32));
                m = fmaxf(m, __shfl_xor(m, 2, 32));
                m = fmaxf(m, __shfl_xor(m, 4, 32));
                m = fmaxf(m, __shfl_xor(m, 8, 32));
                float mn = fmaxf(mrun[ti][i], m);
                corr[i] = __expf(mrun[ti][i] - mn);
                mrun[ti][i] = mn;
                float rs = 0.0f;
#pragma unroll
                for (int tj = 0; tj < 4; ++tj) {
                    float p = __expf(s[ti][tj][i] - mn);
                    s[ti][tj][i] = p;
                    rs += p;
                }
                rs += __shfl_xor(rs, 1, 32);
                rs += __shfl_xor(rs, 2, 32);
                rs += __shfl_xor(rs, 4, 32);
                rs += __shfl_xor(rs, 8, 32);
                lrun[ti][i] = lrun[ti][i] * corr[i] + rs;
            }
#pragma unroll
            for (int nj = 0; nj < 2; ++nj)
#pragma unroll
                for (int i = 0; i < 8; ++i) o[ti][nj][i] *= corr[i];
            // write P (C-layout -> LDS) for A-fragment reload
#pragma unroll
            for (int tj = 0; tj < 4; ++tj)
#pragma unroll
                for (int i = 0; i < 8; ++i)
                    pLds[h][ti * 16 + i + rsel][tj * 16 + laneM] = (bf16_t)s[ti][tj][i];
        }
        // O += P * V   (per-wave LDS, same-wave DS ordering is in-order)
#pragma unroll
        for (int kt = 0; kt < 2; ++kt) {
            v16bf aP[4];
#pragma unroll
            for (int ti = 0; ti < 4; ++ti)
                aP[ti] = frag_a(&pLds[h][ti * 16 + laneM][kt * 32], lane);
#pragma unroll
            for (int nj = 0; nj < 2; ++nj) {
                v16bf bV = frag_b(&vtLds[h][nj * 16 + laneM][kt * 32], lane);
#pragma unroll
                for (int ti = 0; ti < 4; ++ti) o[ti][nj] = wmma_bf(aP[ti], bV, o[ti][nj]);
            }
        }
    }
    // normalize + scatter back through token map (undoes shift + window partition)
#pragma unroll
    for (int ti = 0; ti < 4; ++ti)
#pragma unroll
        for (int nj = 0; nj < 2; ++nj)
#pragma unroll
            for (int i = 0; i < 8; ++i) {
                int tok = q0 + ti * 16 + i + rsel;
                int col = G * 128 + h * 32 + nj * 16 + laneM;
                float val = o[ti][nj][i] / lrun[ti][i];
                xattn[(size_t)tokRow[tok] * 256 + col] = (bf16_t)val;
            }
}

// ---------------------------------------------------------------------------
// Kernel 4: feats = xattn(bf16) @ proj_w + b ; store fp32 feats and reduce
// sum of gelu(feats) over spatial positions into featsS (per batch, channel)
// ---------------------------------------------------------------------------
__global__ __launch_bounds__(256) void gemm_proj(
        const bf16_t* __restrict__ A, const bf16_t* __restrict__ Wt, // Wt: 256x256 N-major
        const float* __restrict__ bias, float* __restrict__ feats,
        float* __restrict__ featsS) {
    const int K = 256;
    __shared__ bf16_t sA[2][128][40];
    __shared__ bf16_t sB[2][128][40];
    __shared__ float  sRed[128];
    int tid = threadIdx.x, lane = tid & 31, wv = tid >> 5;
    int laneM = lane & 15;
    int rowBase = blockIdx.x * 128;
    int c0 = blockIdx.y * 128;
    if (tid < 128) sRed[tid] = 0.0f;

    auto loadA = [&](int k0, v8bf aR[2]) {
#pragma unroll
        for (int it = 0; it < 2; ++it) {
            int g = tid + it * 256, r = g >> 2, kg = g & 3;
            aR[it] = *(const v8bf*)(A + (size_t)(rowBase + r) * 256 + k0 + kg * 8);
        }
    };
    auto storeA = [&](int buf, const v8bf aR[2]) {
#pragma unroll
        for (int it = 0; it < 2; ++it) {
            int g = tid + it * 256, r = g >> 2, kg = g & 3;
            *(v8bf*)&sA[buf][r][kg * 8] = aR[it];
        }
    };
    auto loadB = [&](int k0, v8bf bR[2]) {
#pragma unroll
        for (int it = 0; it < 2; ++it) {
            int g = tid + it * 256, c = g >> 2, kg = g & 3;
            bR[it] = *(const v8bf*)(Wt + (size_t)(c0 + c) * K + k0 + kg * 8);
        }
    };
    auto storeB = [&](int buf, const v8bf bR[2]) {
#pragma unroll
        for (int it = 0; it < 2; ++it) {
            int g = tid + it * 256, c = g >> 2, kg = g & 3;
            *(v8bf*)&sB[buf][c][kg * 8] = bR[it];
        }
    };

    v8f acc[8];
#pragma unroll
    for (int nt = 0; nt < 8; ++nt) acc[nt] = zero_v8f();

    v8bf aR[2], bR[2];
    loadA(0, aR); loadB(0, bR);
    storeA(0, aR); storeB(0, bR);
    __syncthreads();
    for (int kk = 0; kk < 8; ++kk) {
        int cur = kk & 1;
        bool pre = (kk + 1 < 8);
        if (pre) { loadA((kk + 1) * 32, aR); loadB((kk + 1) * 32, bR); }
        v16bf af = frag_a(&sA[cur][wv * 16 + laneM][0], lane);
#pragma unroll
        for (int nt = 0; nt < 8; ++nt) {
            v16bf bfr = frag_b(&sB[cur][nt * 16 + laneM][0], lane);
            acc[nt] = wmma_bf(af, bfr, acc[nt]);
        }
        if (pre) { storeA(1 - cur, aR); storeB(1 - cur, bR); }
        __syncthreads();
    }
    int rsel = (lane & 16) ? 8 : 0;
#pragma unroll
    for (int nt = 0; nt < 8; ++nt) {
#pragma unroll
        for (int i = 0; i < 8; ++i) {
            int row = rowBase + wv * 16 + i + rsel;
            int col = c0 + nt * 16 + laneM;
            float val = acc[nt][i] + bias[col];
            feats[(size_t)row * 256 + col] = val;
            atomicAdd(&sRed[col - c0], gelu_f(val));
        }
    }
    __syncthreads();
    if (tid < 128) {
        int bb = rowBase >> 12;       // 4096 rows per batch, 128-row tiles
        atomicAdd(&featsS[bb * 256 + c0 + tid], sRed[tid]);
    }
}

// ---------------------------------------------------------------------------
// Kernel 5: per-batch SK gate: z = gelu(mean@fc1+b); av = z@fc2+b;
// softmax over the 2 groups per channel -> gw[16][2][128]
// ---------------------------------------------------------------------------
__global__ __launch_bounds__(256) void sk_mix(
        const float* __restrict__ featsS, const float* __restrict__ f1w,
        const float* __restrict__ f1b, const float* __restrict__ f2w,
        const float* __restrict__ f2b, float* __restrict__ gw) {
    __shared__ float sS[256], sZ[128], sAv[256];
    int b = blockIdx.x, t = threadIdx.x;
    sS[t] = featsS[b * 256 + t] * (1.0f / 4096.0f);
    __syncthreads();
    if (t < 128) {
        float acc = f1b[t];
        for (int k = 0; k < 256; ++k) acc += sS[k] * f1w[k * 128 + t];
        sZ[t] = gelu_f(acc);
    }
    __syncthreads();
    float av = f2b[t];
    for (int k = 0; k < 128; ++k) av += sZ[k] * f2w[k * 256 + t];
    sAv[t] = av;
    __syncthreads();
    if (t < 128) {
        float a0 = sAv[t], a1 = sAv[128 + t];
        float m = fmaxf(a0, a1);
        float e0 = __expf(a0 - m), e1 = __expf(a1 - m);
        float inv = 1.0f / (e0 + e1);
        gw[b * 256 + t]       = e0 * inv;
        gw[b * 256 + 128 + t] = e1 * inv;
    }
}

// ---------------------------------------------------------------------------
// Kernel 6: feats_V = sum_g xattn[:, g*128+c] * gw[b][g][c]   (bf16 out)
// ---------------------------------------------------------------------------
__global__ __launch_bounds__(256) void fv_mix(
        const bf16_t* __restrict__ xattn, const float* __restrict__ gw,
        bf16_t* __restrict__ fv) {
    int idx = blockIdx.x * 256 + threadIdx.x;   // 65536*128 elems
    int row = idx >> 7, c = idx & 127;
    int b = row >> 12;
    float x0 = (float)xattn[(size_t)row * 256 + c];
    float x1 = (float)xattn[(size_t)row * 256 + 128 + c];
    fv[(size_t)row * 128 + c] = (bf16_t)(x0 * gw[b * 256 + c] + x1 * gw[b * 256 + 128 + c]);
}

// ---------------------------------------------------------------------------
// Kernel 7: out[b,c,pos] = (fv @ head_w + head_b + feats)  with LDS transpose
// for coalesced NCHW stores (128B lines along pos).
// ---------------------------------------------------------------------------
__global__ __launch_bounds__(256) void gemm_head(
        const bf16_t* __restrict__ A /*65536x128*/, const bf16_t* __restrict__ Wt /*256x128 N-major*/,
        const float* __restrict__ bias, const float* __restrict__ feats,
        float* __restrict__ outp) {
    const int K = 128;
    __shared__ bf16_t sA[2][128][40];
    __shared__ bf16_t sB[2][128][40];
    __shared__ float  sOut[64][132];
    int tid = threadIdx.x, lane = tid & 31, wv = tid >> 5;
    int laneM = lane & 15;
    int rowBase = blockIdx.x * 128;
    int c0 = blockIdx.y * 128;

    auto loadA = [&](int k0, v8bf aR[2]) {
#pragma unroll
        for (int it = 0; it < 2; ++it) {
            int g = tid + it * 256, r = g >> 2, kg = g & 3;
            aR[it] = *(const v8bf*)(A + (size_t)(rowBase + r) * 128 + k0 + kg * 8);
        }
    };
    auto storeA = [&](int buf, const v8bf aR[2]) {
#pragma unroll
        for (int it = 0; it < 2; ++it) {
            int g = tid + it * 256, r = g >> 2, kg = g & 3;
            *(v8bf*)&sA[buf][r][kg * 8] = aR[it];
        }
    };
    auto loadB = [&](int k0, v8bf bR[2]) {
#pragma unroll
        for (int it = 0; it < 2; ++it) {
            int g = tid + it * 256, c = g >> 2, kg = g & 3;
            bR[it] = *(const v8bf*)(Wt + (size_t)(c0 + c) * K + k0 + kg * 8);
        }
    };
    auto storeB = [&](int buf, const v8bf bR[2]) {
#pragma unroll
        for (int it = 0; it < 2; ++it) {
            int g = tid + it * 256, c = g >> 2, kg = g & 3;
            *(v8bf*)&sB[buf][c][kg * 8] = bR[it];
        }
    };

    v8f acc[8];
#pragma unroll
    for (int nt = 0; nt < 8; ++nt) acc[nt] = zero_v8f();

    v8bf aR[2], bR[2];
    loadA(0, aR); loadB(0, bR);
    storeA(0, aR); storeB(0, bR);
    __syncthreads();
    for (int kk = 0; kk < 4; ++kk) {
        int cur = kk & 1;
        bool pre = (kk + 1 < 4);
        if (pre) { loadA((kk + 1) * 32, aR); loadB((kk + 1) * 32, bR); }
        v16bf af = frag_a(&sA[cur][wv * 16 + laneM][0], lane);
#pragma unroll
        for (int nt = 0; nt < 8; ++nt) {
            v16bf bfr = frag_b(&sB[cur][nt * 16 + laneM][0], lane);
            acc[nt] = wmma_bf(af, bfr, acc[nt]);
        }
        if (pre) { storeA(1 - cur, aR); storeB(1 - cur, bR); }
        __syncthreads();
    }

    int rsel = (lane & 16) ? 8 : 0;
    int bb = rowBase >> 12;
    int posBase = rowBase & 4095;
    for (int ch = 0; ch < 2; ++ch) {
        __syncthreads();
#pragma unroll
        for (int nt = ch * 4; nt < ch * 4 + 4; ++nt) {
#pragma unroll
            for (int i = 0; i < 8; ++i) {
                int r = wv * 16 + i + rsel;
                int c = nt * 16 + laneM;
                int row = rowBase + r, col = c0 + c;
                float val = acc[nt][i] + bias[col] + feats[(size_t)row * 256 + col];
                sOut[c - ch * 64][r] = val;
            }
        }
        __syncthreads();
#pragma unroll
        for (int it = 0; it < 8; ++it) {
            int g = tid + it * 256;                 // 0..2047
            int c = g >> 5, r4 = (g & 31) * 4;
            v4f vv = *(const v4f*)&sOut[c][r4];
            *(v4f*)&outp[((size_t)bb * 256 + c0 + ch * 64 + c) * 4096 + posBase + r4] = vv;
        }
    }
}

// ---------------------------------------------------------------------------
// Host launcher
// ---------------------------------------------------------------------------
extern "C" void kernel_launch(void* const* d_in, const int* in_sizes, int n_in,
                              void* d_out, int out_size, void* d_ws, size_t ws_size,
                              hipStream_t stream) {
    (void)in_sizes; (void)n_in; (void)out_size; (void)ws_size;
    const float* x_q    = (const float*)d_in[0];
    const float* x_kv   = (const float*)d_in[1];
    const float* w_q    = (const float*)d_in[2];
    const float* b_q    = (const float*)d_in[3];
    const float* w_kv   = (const float*)d_in[4];
    const float* b_kv   = (const float*)d_in[5];
    const float* rpb0   = (const float*)d_in[6];
    const float* rpb1   = (const float*)d_in[7];
    const float* proj_w = (const float*)d_in[8];
    const float* proj_b = (const float*)d_in[9];
    const float* fc1_w  = (const float*)d_in[10];
    const float* fc1_b  = (const float*)d_in[11];
    const float* fc2_w  = (const float*)d_in[12];
    const float* fc2_b  = (const float*)d_in[13];
    const float* head_w = (const float*)d_in[14];
    const float* head_b = (const float*)d_in[15];
    float* out = (float*)d_out;

    char* ws = (char*)d_ws;
    size_t off = 0;
    auto take = [&](size_t bytes) -> char* {
        char* p = ws + off;
        off = (off + bytes + 255) & ~(size_t)255;
        return p;
    };
    bf16_t* wq_bf  = (bf16_t*)take((size_t)65536 * 2);    // 256x256 -> N x K
    bf16_t* wkv_bf = (bf16_t*)take((size_t)131072 * 2);   // 512 x 256
    bf16_t* pw_bf  = (bf16_t*)take((size_t)65536 * 2);    // 256 x 256
    bf16_t* hw_bf  = (bf16_t*)take((size_t)32768 * 2);    // 256 x 128
    bf16_t* q_bf   = (bf16_t*)take((size_t)65536 * 256 * 2);
    bf16_t* kv_bf  = (bf16_t*)take((size_t)65536 * 512 * 2);
    bf16_t* xat_bf = (bf16_t*)take((size_t)65536 * 256 * 2);
    float*  feats  = (float*) take((size_t)65536 * 256 * 4);
    float*  featsS = (float*) take((size_t)16 * 256 * 4);
    float*  gw     = (float*) take((size_t)16 * 256 * 4);
    bf16_t* fv     = (bf16_t*)take((size_t)65536 * 128 * 2);

    convert_weights<<<512, 256, 0, stream>>>(w_q, w_kv, proj_w, head_w,
                                             wq_bf, wkv_bf, pw_bf, hw_bf);
    gemm_xw_f32a<<<dim3(512, 2), 256, 0, stream>>>(x_q,  wq_bf,  b_q,  q_bf,  256, 256);
    gemm_xw_f32a<<<dim3(512, 4), 256, 0, stream>>>(x_kv, wkv_bf, b_kv, kv_bf, 256, 512);
    win_attn<8, 4><<<1024, 128, 0, stream>>>(q_bf, kv_bf, rpb0, xat_bf);   // 1024 windows
    win_attn<16, 8><<<1024, 128, 0, stream>>>(q_bf, kv_bf, rpb1, xat_bf);  // 256 win x 4 row-blocks
    hipMemsetAsync(featsS, 0, (size_t)16 * 256 * 4, stream);
    gemm_proj<<<dim3(512, 2), 256, 0, stream>>>(xat_bf, pw_bf, proj_b, feats, featsS);
    sk_mix<<<16, 256, 0, stream>>>(featsS, fc1_w, fc1_b, fc2_w, fc2_b, gw);
    fv_mix<<<32768, 256, 0, stream>>>(xat_bf, gw, fv);
    gemm_head<<<dim3(512, 2), 256, 0, stream>>>(fv, hw_bf, head_b, feats, out);
}
